// SANN_fus_25185688223774
// MI455X (gfx1250) — matrix operand order repaired and checked by reading, not compiled, not run on hardware
//
#include <hip/hip_runtime.h>
#include <hip/hip_bf16.h>
#include <math.h>

// ---------------------------------------------------------------------------
// Types / helpers
// ---------------------------------------------------------------------------
typedef __attribute__((ext_vector_type(16))) _Float16 v16h;
typedef __attribute__((ext_vector_type(8)))  _Float16 v8h;
typedef __attribute__((ext_vector_type(8)))  float    v8f;
typedef __attribute__((ext_vector_type(4)))  unsigned int v4u;
typedef __attribute__((ext_vector_type(8)))  int      v8i;
typedef __attribute__((ext_vector_type(4)))  int      v4i;

static inline int  cdiv(long a, long b) { return (int)((a + b - 1) / b); }
static inline long rup(long v, long m)  { return (v + m - 1) / m * m; }

__device__ __forceinline__ float sigmoidf_(float x) { return 1.0f / (1.0f + __expf(-x)); }

// ---------------------------------------------------------------------------
// Tensor Data Mover: async 2D tile (rows x 32 halves) global -> LDS.
// Builds the D# per CDNA5 ISA 8.3/8.4: count=1, type=2 ("image"),
// data_size=1 (2 bytes), tile_dim0=32 elems, tile_dim1=rows,
// tensor_dim0_stride = row stride (elems), tensor_dim1 = rows_rem so the
// TDM zero-fills rows past the matrix edge. Tracked by TENSORcnt.
// ---------------------------------------------------------------------------
__device__ __forceinline__ void tdm_load_2d(const void* gptr, unsigned lds_byte_off,
                                            unsigned rows_rem, unsigned tile_rows,
                                            unsigned stride_elems)
{
#if __has_builtin(__builtin_amdgcn_tensor_load_to_lds)
    unsigned long long ga = (unsigned long long)(size_t)gptr;
    v4u g0;
    g0.x = 1u;                                            // count=1 (valid), user mode
    g0.y = lds_byte_off;                                  // lds_addr
    g0.z = (unsigned)ga;                                  // global_addr[31:0]
    g0.w = (unsigned)((ga >> 32) & 0x01FFFFFFu) | (2u << 30); // addr[56:32] | type=2
    unsigned td0 = stride_elems;                          // tensor_dim0 (always in-bounds)
    unsigned td1 = rows_rem;                              // tensor_dim1 -> OOB rows = 0
    v8i g1;
    g1[0] = (int)(1u << 16);                              // wg_mask=0 | data_size=1 (2B)
    g1[1] = (int)((td0 & 0xFFFFu) << 16);                 // tensor_dim0[15:0]
    g1[2] = (int)((td0 >> 16) | ((td1 & 0xFFFFu) << 16)); // td0[31:16] | td1[15:0]
    g1[3] = (int)((td1 >> 16) | (32u << 16));             // td1[31:16] | tile_dim0=32
    g1[4] = (int)(tile_rows & 0xFFFFu);                   // tile_dim1 | tile_dim2=0
    g1[5] = (int)stride_elems;                            // tensor_dim0_stride[31:0]
    g1[6] = 0;                                            // stride[47:32] | dim1_stride lo
    g1[7] = 0;
    v4i gz = {0, 0, 0, 0};
#if defined(__clang_major__) && __clang_major__ >= 23
    v8i gz8 = {0, 0, 0, 0, 0, 0, 0, 0};
    __builtin_amdgcn_tensor_load_to_lds(g0, g1, gz, gz, gz8, 0);
#else
    __builtin_amdgcn_tensor_load_to_lds(g0, g1, gz, gz, 0);
#endif
#else
    (void)gptr; (void)lds_byte_off; (void)rows_rem; (void)tile_rows; (void)stride_elems;
#endif
}

// ---------------------------------------------------------------------------
// WMMA GEMM: C[M,N] = alpha * A[M,Kp] * W[N,Kp]^T + bias[N]
//   A, W : fp16 row-major, rows padded with zeros to Kp (Kp % 32 == 0)
//   C    : fp32 row-major, leading dim N
//   Batched via blockIdx.z with element strides (halves for A/W, floats for C)
// Block = 256 threads = 8 wave32 (4x2); block tile 128x64; each wave 32x32
// -> 4 x v_wmma_f32_16x16x32_f16 per K-step (2 A-frags x 2 B-frags).
// Tiles staged into LDS by the TDM (wave 0 issues, TENSORcnt drains); falls
// back to cooperative b128 staging where the TDM builtin is unavailable.
// ---------------------------------------------------------------------------
__global__ __launch_bounds__(256)
void k_gemm_f16(const _Float16* __restrict__ A, const _Float16* __restrict__ W,
                const float* __restrict__ bias, float* __restrict__ C,
                int M, int N, int Kp, float alpha,
                long strideA, long strideW, long strideC)
{
    __shared__ _Float16 smem[(128 + 64) * 32];   // As @ 0 (8KB), Bs @ 8KB (4KB)
    _Float16* As = smem;
    _Float16* Bs = smem + 128 * 32;

    const int tid  = threadIdx.x;
    const int wave = tid >> 5;          // 0..7
    const int lane = tid & 31;
    const int wrow = wave & 3;          // 32-row slab within 128
    const int wcol = wave >> 2;         // 32-col slab within 64

    const long z = blockIdx.z;
    A += z * strideA;  W += z * strideW;  C += z * strideC;

    const int m0 = blockIdx.y * 128;
    const int n0 = blockIdx.x * 64;

    const int lhalf = lane & 15;
    const int lhi   = lane >> 4;        // 0/1

    const unsigned ldsA = (unsigned)(size_t)(void*)As;
    const unsigned ldsB = (unsigned)(size_t)(void*)Bs;

    v8f acc00 = {}; v8f acc01 = {}; v8f acc10 = {}; v8f acc11 = {};

    for (int k0 = 0; k0 < Kp; k0 += 32) {
#if __has_builtin(__builtin_amdgcn_tensor_load_to_lds)
        if (wave == 0) {
            tdm_load_2d(A + (long)m0 * Kp + k0, ldsA, (unsigned)(M - m0), 128u,
                        (unsigned)Kp);
            tdm_load_2d(W + (long)n0 * Kp + k0, ldsB, (unsigned)(N - n0), 64u,
                        (unsigned)Kp);
#if __has_builtin(__builtin_amdgcn_s_wait_tensorcnt)
            __builtin_amdgcn_s_wait_tensorcnt(0);
#endif
        }
        __syncthreads();
#else
        // cooperative staging fallback: A 128x32 (2 chunks/thread), B 64x32
        {
            int q = tid;
#pragma unroll
            for (int rep = 0; rep < 2; ++rep, q += 256) {
                int r = q >> 2, cc = (q & 3) * 8;
                int gm = m0 + r;
                v8h v = {};
                if (gm < M) v = *(const v8h*)(A + (long)gm * Kp + k0 + cc);
                *(v8h*)(As + r * 32 + cc) = v;
            }
            int r = tid >> 2, cc = (tid & 3) * 8;
            int gn = n0 + r;
            v8h v = {};
            if (gn < N) v = *(const v8h*)(W + (long)gn * Kp + k0 + cc);
            *(v8h*)(Bs + r * 32 + cc) = v;
        }
        __syncthreads();
#endif

        // A fragments (16x32 each): lane<16 rows M=lhalf with K 0..15;
        // lanes>=16 rows M=lhalf with K 16..31
        const int am0   = (wrow * 32 + lhalf) * 32;
        const int am1   = am0 + 16 * 32;
        const int abase = lhi * 16;
        v8h  a0lo = *(const v8h*)(As + am0 + abase);
        v8h  a0hi = *(const v8h*)(As + am0 + abase + 8);
        v16h af0  = __builtin_shufflevector(a0lo, a0hi,
                        0,1,2,3,4,5,6,7,8,9,10,11,12,13,14,15);
        v8h  a1lo = *(const v8h*)(As + am1 + abase);
        v8h  a1hi = *(const v8h*)(As + am1 + abase + 8);
        v16h af1  = __builtin_shufflevector(a1lo, a1hi,
                        0,1,2,3,4,5,6,7,8,9,10,11,12,13,14,15);

        // B fragments: lane -> N=lhalf, K = lhi*16 + i (16 contiguous halves)
        const int boff = (wcol * 32 + lhalf) * 32 + lhi * 16;
        v16h bf0 = *(const v16h*)(Bs + boff);
        v16h bf1 = *(const v16h*)(Bs + boff + 16 * 32);

        acc00 = __builtin_amdgcn_wmma_f32_16x16x32_f16(false, af0, false, bf0,
                                                       (short)0, acc00, false, false);
        acc01 = __builtin_amdgcn_wmma_f32_16x16x32_f16(false, af0, false, bf1,
                                                       (short)0, acc01, false, false);
        acc10 = __builtin_amdgcn_wmma_f32_16x16x32_f16(false, af1, false, bf0,
                                                       (short)0, acc10, false, false);
        acc11 = __builtin_amdgcn_wmma_f32_16x16x32_f16(false, af1, false, bf1,
                                                       (short)0, acc11, false, false);
        __syncthreads();
    }

    // epilogue: lane -> col N = lhalf within tile; reg r -> row M = r + 8*lhi
    const int cm = m0 + wrow * 32;
    const int cn = n0 + wcol * 32 + lhalf;
#pragma unroll
    for (int r = 0; r < 8; ++r) {
        int gm0 = cm + lhi * 8 + r;
        int gm1 = gm0 + 16;
        if (gm0 < M) {
            float* row = C + (long)gm0 * N;
            if (cn < N)      row[cn]      = alpha * acc00[r] + (bias ? bias[cn]      : 0.f);
            if (cn + 16 < N) row[cn + 16] = alpha * acc01[r] + (bias ? bias[cn + 16] : 0.f);
        }
        if (gm1 < M) {
            float* row = C + (long)gm1 * N;
            if (cn < N)      row[cn]      = alpha * acc10[r] + (bias ? bias[cn]      : 0.f);
            if (cn + 16 < N) row[cn + 16] = alpha * acc11[r] + (bias ? bias[cn + 16] : 0.f);
        }
    }
}

// ---------------------------------------------------------------------------
// Glue kernels
// ---------------------------------------------------------------------------
__global__ void k_avgpool8(const float* __restrict__ x, float* __restrict__ y, int total)
{
    int i = blockIdx.x * 256 + threadIdx.x;
    if (i >= total) return;
    int ox = i % 7; int t = i / 7; int oy = t % 7; t /= 7; int c = t % 64; int b = t / 64;
    const float* p = x + (((long)b * 64 + c) * 56 + oy * 8) * 56 + ox * 8;
    float s = 0.f;
    for (int r = 0; r < 8; ++r)
        for (int q = 0; q < 8; ++q) s += p[r * 56 + q];
    y[i] = s * (1.0f / 64.0f);
}

// im2col 3x3 pad1 on 7x7 grid; dst[b*49+pix][c*9+kh*3+kw] fp16 (Kp == C*9)
__global__ void k_im2col3(const float* __restrict__ src, _Float16* __restrict__ dst,
                          int C, long sb, long sc, long sp, long total)
{
    long i = blockIdx.x * 256L + threadIdx.x;
    if (i >= total) return;
    int kw = (int)(i % 3); long t = i / 3;
    int kh = (int)(t % 3); t /= 3;
    int c  = (int)(t % C); t /= C;
    int p  = (int)(t % 49); int b = (int)(t / 49);
    int oy = p / 7, ox = p % 7;
    int iy = oy + kh - 1, ix = ox + kw - 1;
    float v = 0.f;
    if (iy >= 0 && iy < 7 && ix >= 0 && ix < 7)
        v = src[b * sb + c * sc + (iy * 7 + ix) * sp];
    dst[i] = (_Float16)v;
}

// fp32 [rows][K] -> fp16 [rows][Kp] zero padded
__global__ void k_cvt_pad(const float* __restrict__ s, _Float16* __restrict__ d,
                          long total, int K, int Kp)
{
    long i = blockIdx.x * 256L + threadIdx.x;
    if (i >= total) return;
    int k = (int)(i % Kp); long r = i / Kp;
    d[i] = (k < K) ? (_Float16)s[r * (long)K + k] : (_Float16)0.f;
}

__global__ void k_gelu_f16(const float* __restrict__ s, _Float16* __restrict__ d,
                           long total, int K, int Kp)
{
    long i = blockIdx.x * 256L + threadIdx.x;
    if (i >= total) return;
    int k = (int)(i % Kp); long r = i / Kp;
    float v = 0.f;
    if (k < K) {
        float x = s[r * (long)K + k];
        v = 0.5f * x * (1.0f + erff(x * 0.70710678118654752f));
    }
    d[i] = (_Float16)v;
}

__global__ void k_layernorm_f16(const float* __restrict__ x, const float* __restrict__ g,
                                const float* __restrict__ b, _Float16* __restrict__ y,
                                int C, int Cp)
{
    long row = blockIdx.x;
    const float* p = x + row * (long)C;
    __shared__ float red[256];
    float s = 0.f;
    for (int j = threadIdx.x; j < C; j += 256) s += p[j];
    red[threadIdx.x] = s; __syncthreads();
    for (int t = 128; t > 0; t >>= 1) {
        if (threadIdx.x < t) red[threadIdx.x] += red[threadIdx.x + t];
        __syncthreads();
    }
    float mean = red[0] / C;
    __syncthreads();
    float v = 0.f;
    for (int j = threadIdx.x; j < C; j += 256) { float d = p[j] - mean; v += d * d; }
    red[threadIdx.x] = v; __syncthreads();
    for (int t = 128; t > 0; t >>= 1) {
        if (threadIdx.x < t) red[threadIdx.x] += red[threadIdx.x + t];
        __syncthreads();
    }
    float rstd = rsqrtf(red[0] / C + 1e-6f);
    _Float16* yr = y + row * (long)Cp;
    for (int j = threadIdx.x; j < Cp; j += 256)
        yr[j] = (j < C) ? (_Float16)((p[j] - mean) * rstd * g[j] + b[j]) : (_Float16)0.f;
}

// row softmax with per-head 0/1 mask applied to logits; grid (row, head)
__global__ void k_softmax_mask(float* __restrict__ attn, int Ntok,
                               const int* __restrict__ mask, int H)
{
    long bh = blockIdx.y;
    float* p = attn + (bh * Ntok + blockIdx.x) * (long)Ntok;
    float mval = (float)mask[bh % H];
    __shared__ float red[256];
    float mx = -3.402823e38f;
    for (int j = threadIdx.x; j < Ntok; j += 256) { p[j] *= mval; mx = fmaxf(mx, p[j]); }
    red[threadIdx.x] = mx; __syncthreads();
    for (int t = 128; t > 0; t >>= 1) {
        if (threadIdx.x < t) red[threadIdx.x] = fmaxf(red[threadIdx.x], red[threadIdx.x + t]);
        __syncthreads();
    }
    mx = red[0]; __syncthreads();
    float s = 0.f;
    for (int j = threadIdx.x; j < Ntok; j += 256) { float e = __expf(p[j] - mx); p[j] = e; s += e; }
    red[threadIdx.x] = s; __syncthreads();
    for (int t = 128; t > 0; t >>= 1) {
        if (threadIdx.x < t) red[threadIdx.x] += red[threadIdx.x + t];
        __syncthreads();
    }
    float inv = 1.0f / red[0];
    for (int j = threadIdx.x; j < Ntok; j += 256) p[j] *= inv;
}

__global__ void k_split_qk(const float* __restrict__ qkv, _Float16* __restrict__ q,
                           _Float16* __restrict__ k, int Ntok, int H, int D,
                           int Dp, long total)
{
    long i = blockIdx.x * 256L + threadIdx.x;
    if (i >= total) return;
    int d = (int)(i % Dp); long t = i / Dp;
    int n = (int)(t % Ntok); t /= Ntok;
    int h = (int)(t % H); int b = (int)(t / H);
    int C = H * D;
    float qv = 0.f, kv = 0.f;
    if (d < D) {
        const float* base = qkv + ((long)b * Ntok + n) * 3 * C + h * D + d;
        qv = base[0];
        kv = base[C];
    }
    q[i] = (_Float16)qv;
    k[i] = (_Float16)kv;
}

__global__ void k_split_vT(const float* __restrict__ qkv, _Float16* __restrict__ vT,
                           int Ntok, int H, int D, int Np, long total)
{
    long i = blockIdx.x * 256L + threadIdx.x;
    if (i >= total) return;
    int n = (int)(i % Np); long t = i / Np;
    int d = (int)(t % D); t /= D;
    int h = (int)(t % H); int b = (int)(t / H);
    int C = H * D;
    float v = 0.f;
    if (n < Ntok) v = qkv[((long)b * Ntok + n) * 3 * C + 2 * C + h * D + d];
    vT[i] = (_Float16)v;
}

__global__ void k_merge_heads(const float* __restrict__ ho, float* __restrict__ out,
                              int Ntok, int H, int D, long total)
{
    long i = blockIdx.x * 256L + threadIdx.x;
    if (i >= total) return;
    int d = (int)(i % D); long t = i / D;
    int h = (int)(t % H); t /= H;
    int n = (int)(t % Ntok); int b = (int)(t / Ntok);
    out[((long)b * Ntok + n) * (H * D) + h * D + d] =
        ho[(((long)b * H + h) * Ntok + n) * D + d];
}

__global__ void k_add(float* __restrict__ x, const float* __restrict__ y, long n)
{
    long i = blockIdx.x * 256L + threadIdx.x;
    if (i < n) x[i] += y[i];
}

__global__ void k_copy(float* __restrict__ d, const float* __restrict__ s, long n)
{
    long i = blockIdx.x * 256L + threadIdx.x;
    if (i < n) d[i] = s[i];
}

__global__ void k_chan_mean(const float* __restrict__ fm, float* __restrict__ out,
                            int B, int C, int P)
{
    int i = blockIdx.x * 256 + threadIdx.x;
    if (i >= B * C) return;
    int c = i % C, b = i / C;
    const float* p = fm + (long)b * P * C + c;
    float s = 0.f;
    for (int q = 0; q < P; ++q) s += p[(long)q * C];
    out[i] = s / P;
}

__global__ void k_chan_max(const float* __restrict__ fm, float* __restrict__ out,
                           int B, int C, int P)
{
    int i = blockIdx.x * 256 + threadIdx.x;
    if (i >= B * C) return;
    int c = i % C, b = i / C;
    const float* p = fm + (long)b * P * C + c;
    float s = -3.402823e38f;
    for (int q = 0; q < P; ++q) s = fmaxf(s, p[(long)q * C]);
    out[i] = s;
}

__global__ void k_mlp_relu(const float* __restrict__ v, const float* __restrict__ w,
                           const float* __restrict__ bias, float* __restrict__ out,
                           int B, int In, int Out)
{
    int i = blockIdx.x * 256 + threadIdx.x;
    if (i >= B * Out) return;
    int j = i % Out, b = i / Out;
    const float* vb = v + (long)b * In;
    const float* wr = w + (long)j * In;
    float s = bias[j];
    for (int c = 0; c < In; ++c) s += vb[c] * wr[c];
    out[i] = fmaxf(s, 0.f);
}

__global__ void k_mlp_sigmoid(const float* __restrict__ v, const float* __restrict__ w,
                              const float* __restrict__ bias, float* __restrict__ out,
                              int B, int In, int Out)
{
    int i = blockIdx.x * 256 + threadIdx.x;
    if (i >= B * Out) return;
    int j = i % Out, b = i / Out;
    const float* vb = v + (long)b * In;
    const float* wr = w + (long)j * In;
    float s = bias[j];
    for (int c = 0; c < In; ++c) s += vb[c] * wr[c];
    out[i] = sigmoidf_(s);
}

__global__ void k_scale_chan(float* __restrict__ fm, const float* __restrict__ s,
                             long total, int P, int C)
{
    long i = blockIdx.x * 256L + threadIdx.x;
    if (i >= total) return;
    int c = (int)(i % C);
    long b = i / ((long)P * C);
    fm[i] *= s[b * C + c];
}

__global__ void k_scale_spatial(float* __restrict__ fm, const float* __restrict__ g,
                                long total, int C)
{
    long i = blockIdx.x * 256L + threadIdx.x;
    if (i >= total) return;
    fm[i] *= g[i / C];
}

__global__ void k_bnrelu(float* __restrict__ h, const float* __restrict__ g,
                         const float* __restrict__ bb, long total, int C)
{
    long i = blockIdx.x * 256L + threadIdx.x;
    if (i >= total) return;
    int c = (int)(i % C);
    h[i] = fmaxf(h[i] * (g[c] * rsqrtf(1.0f + 1e-5f)) + bb[c], 0.f);
}

// 1x1 conv C->1 + sigmoid, rows = B*49
__global__ void k_gate1(const float* __restrict__ h, const float* __restrict__ w,
                        const float* __restrict__ b2, float* __restrict__ gate,
                        int rows, int C)
{
    int i = blockIdx.x * 256 + threadIdx.x;
    if (i >= rows) return;
    const float* hr = h + (long)i * C;
    float s = b2[0];
    for (int j = 0; j < C; ++j) s += hr[j] * w[j];
    gate[i] = sigmoidf_(s);
}

// 3x3 conv C->1 pad1 on 7x7 + sigmoid; h layout [B][49][C], w [C][3][3]
__global__ void k_gate3(const float* __restrict__ h, const float* __restrict__ w,
                        const float* __restrict__ b2, float* __restrict__ gate,
                        int B, int C)
{
    int i = blockIdx.x * 256 + threadIdx.x;
    if (i >= B * 49) return;
    int p = i % 49, b = i / 49;
    int oy = p / 7, ox = p % 7;
    float s = b2[0];
    for (int kh = 0; kh < 3; ++kh) {
        int iy = oy + kh - 1; if (iy < 0 || iy >= 7) continue;
        for (int kw = 0; kw < 3; ++kw) {
            int ix = ox + kw - 1; if (ix < 0 || ix >= 7) continue;
            const float* hr = h + ((long)b * 49 + iy * 7 + ix) * C;
            for (int j = 0; j < C; ++j) s += hr[j] * w[j * 9 + kh * 3 + kw];
        }
    }
    gate[i] = sigmoidf_(s);
}

// quadrant-tile the four [B][49][768] maps into token-major and channel-major fmaps
__global__ void k_tile_fmap(const float* __restrict__ f1, const float* __restrict__ f2,
                            const float* __restrict__ f3, const float* __restrict__ f4,
                            float* __restrict__ tok, float* __restrict__ chan,
                            int C, long total)
{
    long i = blockIdx.x * 256L + threadIdx.x;
    if (i >= total) return;
    int c = (int)(i % C); long t = i / C;
    int s = (int)(t % 196); int b = (int)(t / 196);
    int y = s / 14, xx = s % 14;
    const float* src; int p;
    if (xx < 7) { if (y < 7) { src = f1; p = y * 7 + xx; }
                  else       { src = f2; p = (y - 7) * 7 + xx; } }
    else        { if (y < 7) { src = f3; p = y * 7 + (xx - 7); }
                  else       { src = f4; p = (y - 7) * 7 + (xx - 7); } }
    float v = src[((long)b * 49 + p) * C + c];
    tok[i] = v;
    chan[((long)b * C + c) * 196 + s] = v;
}

// out[b][c][s] = 0.5*(a1_tok[b][s][c] + a2_chan[b][c][s])
__global__ void k_combine(const float* __restrict__ a1tok, const float* __restrict__ a2chan,
                          float* __restrict__ out, long total, int S, int C)
{
    long i = blockIdx.x * 256L + threadIdx.x;
    if (i >= total) return;
    int s = (int)(i % S); long t = i / S;
    int c = (int)(t % C); int b = (int)(t / C);
    out[i] = 0.5f * (a1tok[((long)b * S + s) * C + c] + a2chan[i]);
}

// ---------------------------------------------------------------------------
// Host-side orchestration
// ---------------------------------------------------------------------------
struct Arena {
    char* base; size_t off;
    template <typename T> T* take(size_t n) {
        T* p = (T*)(base + off);
        off = (off + n * sizeof(T) + 255) & ~(size_t)255;
        return p;
    }
};

struct BlkW {
    const float *ln1_g, *ln1_b, *qkv_w, *qkv_b, *proj_w, *proj_b;
    const float *ln2_g, *ln2_b, *fc1_w, *fc1_b, *fc2_w, *fc2_b;
};

static void gemm(const _Float16* A, const _Float16* W, const float* bias, float* C,
                 int M, int N, int Kp, float alpha, int Z,
                 long sA, long sW, long sC, hipStream_t st)
{
    dim3 g(cdiv(N, 64), cdiv(M, 128), Z);
    k_gemm_f16<<<g, 256, 0, st>>>(A, W, bias, C, M, N, Kp, alpha, sA, sW, sC);
}

static void cvt(const float* s, _Float16* d, long rows, int K, int Kp, hipStream_t st)
{
    long tot = rows * (long)Kp;
    k_cvt_pad<<<cdiv(tot, 256), 256, 0, st>>>(s, d, tot, K, Kp);
}

// ViT block with per-head logit masking; x = residual stream [B*Ntok][C], mutated.
static void run_block(float* x, const BlkW& w, int B, int Ntok, int C, int H,
                      const int* mask, Arena ar, hipStream_t st)
{
    const int BN = B * Ntok;
    const int D  = C / H;
    const int Cp = (int)rup(C, 32);
    const int Dp = (int)rup(D, 32);
    const int Np = (int)rup(Ntok, 32);
    const int F  = 4 * C;
    const int Fp = (int)rup(F, 32);
    const float scale = 1.0f / sqrtf((float)D);

    // --- attention half ---
    _Float16* lnh = ar.take<_Float16>((size_t)BN * Cp);
    k_layernorm_f16<<<BN, 256, 0, st>>>(x, w.ln1_g, w.ln1_b, lnh, C, Cp);

    _Float16* qkvw = ar.take<_Float16>((size_t)3 * C * Cp);
    cvt(w.qkv_w, qkvw, 3 * C, C, Cp, st);
    float* qkv = ar.take<float>((size_t)BN * 3 * C);
    gemm(lnh, qkvw, w.qkv_b, qkv, BN, 3 * C, Cp, 1.f, 1, 0, 0, 0, st);

    _Float16* qh = ar.take<_Float16>((size_t)B * H * Ntok * Dp);
    _Float16* kh = ar.take<_Float16>((size_t)B * H * Ntok * Dp);
    _Float16* vT = ar.take<_Float16>((size_t)B * H * D * Np);
    {
        long tot = (long)B * H * Ntok * Dp;
        k_split_qk<<<cdiv(tot, 256), 256, 0, st>>>(qkv, qh, kh, Ntok, H, D, Dp, tot);
    }
    {
        long tot = (long)B * H * D * Np;
        k_split_vT<<<cdiv(tot, 256), 256, 0, st>>>(qkv, vT, Ntok, H, D, Np, tot);
    }

    float*    attn_b = ar.take<float>((size_t)H * Ntok * Ntok);
    _Float16* attn_h = ar.take<_Float16>((size_t)H * Ntok * Np);
    float*    ho     = ar.take<float>((size_t)B * H * Ntok * D);
    for (int b = 0; b < B; ++b) {
        const _Float16* qb = qh + (size_t)b * H * Ntok * Dp;
        const _Float16* kb = kh + (size_t)b * H * Ntok * Dp;
        gemm(qb, kb, nullptr, attn_b, Ntok, Ntok, Dp, scale, H,
             (long)Ntok * Dp, (long)Ntok * Dp, (long)Ntok * Ntok, st);
        k_softmax_mask<<<dim3(Ntok, H), 256, 0, st>>>(attn_b, Ntok, mask, H);
        cvt(attn_b, attn_h, (long)H * Ntok, Ntok, Np, st);
        gemm(attn_h, vT + (size_t)b * H * D * Np, nullptr,
             ho + (size_t)b * H * Ntok * D,
             Ntok, D, Np, 1.f, H, (long)Ntok * Np, (long)D * Np, (long)Ntok * D, st);
    }

    float* merged = ar.take<float>((size_t)BN * C);
    {
        long tot = (long)BN * C;
        k_merge_heads<<<cdiv(tot, 256), 256, 0, st>>>(ho, merged, Ntok, H, D, tot);
    }
    _Float16* mh = ar.take<_Float16>((size_t)BN * Cp);
    cvt(merged, mh, BN, C, Cp, st);
    _Float16* pw = ar.take<_Float16>((size_t)C * Cp);
    cvt(w.proj_w, pw, C, C, Cp, st);
    float* tmp = ar.take<float>((size_t)BN * C);
    gemm(mh, pw, w.proj_b, tmp, BN, C, Cp, 1.f, 1, 0, 0, 0, st);
    k_add<<<cdiv((long)BN * C, 256), 256, 0, st>>>(x, tmp, (long)BN * C);

    // --- MLP half ---
    _Float16* ln2h = ar.take<_Float16>((size_t)BN * Cp);
    k_layernorm_f16<<<BN, 256, 0, st>>>(x, w.ln2_g, w.ln2_b, ln2h, C, Cp);
    _Float16* f1w = ar.take<_Float16>((size_t)F * Cp);
    cvt(w.fc1_w, f1w, F, C, Cp, st);
    float* h1 = ar.take<float>((size_t)BN * F);
    gemm(ln2h, f1w, w.fc1_b, h1, BN, F, Cp, 1.f, 1, 0, 0, 0, st);
    _Float16* gh = ar.take<_Float16>((size_t)BN * Fp);
    {
        long tot = (long)BN * Fp;
        k_gelu_f16<<<cdiv(tot, 256), 256, 0, st>>>(h1, gh, tot, F, Fp);
    }
    _Float16* f2w = ar.take<_Float16>((size_t)C * Fp);
    cvt(w.fc2_w, f2w, C, F, Fp, st);
    gemm(gh, f2w, w.fc2_b, tmp, BN, C, Fp, 1.f, 1, 0, 0, 0, st);
    k_add<<<cdiv((long)BN * C, 256), 256, 0, st>>>(x, tmp, (long)BN * C);
}

// ---------------------------------------------------------------------------
// Entry point
// ---------------------------------------------------------------------------
extern "C" void kernel_launch(void* const* d_in, const int* in_sizes, int n_in,
                              void* d_out, int out_size, void* d_ws, size_t ws_size,
                              hipStream_t stream)
{
    (void)in_sizes; (void)n_in; (void)out_size; (void)ws_size;
    const int B = 16;

    const float* x = (const float*)d_in[0];
    const float* cw[4] = {(const float*)d_in[1], (const float*)d_in[3],
                          (const float*)d_in[5], (const float*)d_in[7]};
    const float* cb[4] = {(const float*)d_in[2], (const float*)d_in[4],
                          (const float*)d_in[6], (const float*)d_in[8]};
    const float* sp1w1 = (const float*)d_in[9],  *sp1b1 = (const float*)d_in[10];
    const float* sp1g  = (const float*)d_in[11], *sp1bb = (const float*)d_in[12];
    const float* sp1w2 = (const float*)d_in[13], *sp1b2 = (const float*)d_in[14];
    const float* sp3w1 = (const float*)d_in[15], *sp3b1 = (const float*)d_in[16];
    const float* sp3g  = (const float*)d_in[17], *sp3bb = (const float*)d_in[18];
    const float* sp3w2 = (const float*)d_in[19], *sp3b2 = (const float*)d_in[20];
    const float* caw1 = (const float*)d_in[21], *cab1 = (const float*)d_in[22];
    const float* caw2 = (const float*)d_in[23], *cab2 = (const float*)d_in[24];
    const float* cmw1 = (const float*)d_in[25], *cmb1 = (const float*)d_in[26];
    const float* cmw2 = (const float*)d_in[27], *cmb2 = (const float*)d_in[28];
    BlkW sa = {(const float*)d_in[29], (const float*)d_in[30], (const float*)d_in[31],
               (const float*)d_in[32], (const float*)d_in[33], (const float*)d_in[34],
               (const float*)d_in[35], (const float*)d_in[36], (const float*)d_in[37],
               (const float*)d_in[38], (const float*)d_in[39], (const float*)d_in[40]};
    BlkW fa = {(const float*)d_in[41], (const float*)d_in[42], (const float*)d_in[43],
               (const float*)d_in[44], (const float*)d_in[45], (const float*)d_in[46],
               (const float*)d_in[47], (const float*)d_in[48], (const float*)d_in[49],
               (const float*)d_in[50], (const float*)d_in[51], (const float*)d_in[52]};
    const int* msa = (const int*)d_in[53];
    const int* mfa = (const int*)d_in[54];

    // persistent region
    char* ws = (char*)d_ws;
    size_t off = 0;
    auto ptake = [&](size_t bytes) -> void* {
        void* p = ws + off; off = (off + bytes + 255) & ~(size_t)255; return p;
    };
    const size_t FMN = (size_t)B * 196 * 768;
    float* fm_tok  = (float*)ptake(FMN * sizeof(float));   // [b][s][c]
    float* fm_chan = (float*)ptake(FMN * sizeof(float));   // [b][c][s]
    float* xa1     = (float*)ptake(FMN * sizeof(float));   // SA1 residual stream
    Arena ar{ws + off, 0};

    // ---- phase 0: pool, convs, gates, fmap ----
    float* x7 = ar.take<float>((size_t)B * 64 * 49);
    k_avgpool8<<<cdiv(B * 64 * 49, 256), 256, 0, stream>>>(x, x7, B * 64 * 49);

    _Float16* pat1 = ar.take<_Float16>((size_t)784 * 576);
    {
        long tot = (long)784 * 576;
        k_im2col3<<<cdiv(tot, 256), 256, 0, stream>>>(x7, pat1, 64,
                                                      (long)64 * 49, 49L, 1L, tot);
    }
    _Float16* wch = ar.take<_Float16>((size_t)768 * 576);
    float* conv[4];
    for (int i = 0; i < 4; ++i) {
        conv[i] = ar.take<float>((size_t)784 * 768);
        cvt(cw[i], wch, 768, 576, 576, stream);
        gemm(pat1, wch, cb[i], conv[i], 784, 768, 576, 1.f, 1, 0, 0, 0, stream);
    }

    // SE channel-avg gate on fm1
    float* vred = ar.take<float>((size_t)B * 768);
    float* hid  = ar.take<float>((size_t)B * 192);
    float* gvec = ar.take<float>((size_t)B * 768);
    k_chan_mean<<<cdiv(B * 768, 256), 256, 0, stream>>>(conv[0], vred, B, 768, 49);
    k_mlp_relu<<<cdiv(B * 192, 256), 256, 0, stream>>>(vred, caw1, cab1, hid, B, 768, 192);
    k_mlp_sigmoid<<<cdiv(B * 768, 256), 256, 0, stream>>>(hid, caw2, cab2, gvec, B, 192, 768);
    k_scale_chan<<<cdiv((long)784 * 768, 256), 256, 0, stream>>>(conv[0], gvec,
                                                                 (long)784 * 768, 49, 768);

    // 1x1 spatial gate on fm2 (conv 768->192 via WMMA, 192->1 direct)
    _Float16* fm2h = ar.take<_Float16>((size_t)784 * 768);
    cvt(conv[1], fm2h, 784, 768, 768, stream);
    _Float16* w1h = ar.take<_Float16>((size_t)192 * 768);
    cvt(sp1w1, w1h, 192, 768, 768, stream);
    float* g1h = ar.take<float>((size_t)784 * 192);
    gemm(fm2h, w1h, sp1b1, g1h, 784, 192, 768, 1.f, 1, 0, 0, 0, stream);
    k_bnrelu<<<cdiv((long)784 * 192, 256), 256, 0, stream>>>(g1h, sp1g, sp1bb,
                                                             (long)784 * 192, 192);
    float* gate1 = ar.take<float>(1024);
    k_gate1<<<cdiv(784, 256), 256, 0, stream>>>(g1h, sp1w2, sp1b2, gate1, 784, 192);
    k_scale_spatial<<<cdiv((long)784 * 768, 256), 256, 0, stream>>>(conv[1], gate1,
                                                                    (long)784 * 768, 768);

    // SE channel-max gate (from gated fm1) on fm3
    k_chan_max<<<cdiv(B * 768, 256), 256, 0, stream>>>(conv[0], vred, B, 768, 49);
    k_mlp_relu<<<cdiv(B * 192, 256), 256, 0, stream>>>(vred, cmw1, cmb1, hid, B, 768, 192);
    k_mlp_sigmoid<<<cdiv(B * 768, 256), 256, 0, stream>>>(hid, cmw2, cmb2, gvec, B, 192, 768);
    k_scale_chan<<<cdiv((long)784 * 768, 256), 256, 0, stream>>>(conv[2], gvec,
                                                                 (long)784 * 768, 49, 768);

    // 3x3 spatial gate on fm4 (im2col + WMMA 6912->192, then 192->1 3x3 direct)
    _Float16* pat3 = ar.take<_Float16>((size_t)784 * 6912);
    {
        long tot = (long)784 * 6912;
        k_im2col3<<<cdiv(tot, 256), 256, 0, stream>>>(conv[3], pat3, 768,
                                                      (long)49 * 768, 1L, 768L, tot);
    }
    _Float16* w3h = ar.take<_Float16>((size_t)192 * 6912);
    cvt(sp3w1, w3h, 192, 6912, 6912, stream);
    float* g3h = ar.take<float>((size_t)784 * 192);
    gemm(pat3, w3h, sp3b1, g3h, 784, 192, 6912, 1.f, 1, 0, 0, 0, stream);
    k_bnrelu<<<cdiv((long)784 * 192, 256), 256, 0, stream>>>(g3h, sp3g, sp3bb,
                                                             (long)784 * 192, 192);
    float* gate3 = ar.take<float>(1024);
    k_gate3<<<cdiv(784, 256), 256, 0, stream>>>(g3h, sp3w2, sp3b2, gate3, B, 192);
    k_scale_spatial<<<cdiv((long)784 * 768, 256), 256, 0, stream>>>(conv[3], gate3,
                                                                    (long)784 * 768, 768);

    // tile quadrants into fmap (token-major + channel-major)
    {
        long tot = (long)FMN;
        k_tile_fmap<<<cdiv(tot, 256), 256, 0, stream>>>(conv[0], conv[1], conv[2], conv[3],
                                                        fm_tok, fm_chan, 768, tot);
    }

    // ---- phase 1: SA1 (196 tokens, dim 768, 12 heads) on a copy of fm_tok ----
    ar.off = 0;
    k_copy<<<cdiv((long)FMN, 256), 256, 0, stream>>>(xa1, fm_tok, (long)FMN);
    run_block(xa1, sa, B, 196, 768, 12, msa, ar, stream);

    // ---- phase 2: FA1 (768 tokens, dim 196, 14 heads) in-place on fm_chan ----
    ar.off = 0;
    run_block(fm_chan, fa, B, 768, 196, 14, mfa, ar, stream);

    // ---- combine: out[b][c][y][x] = 0.5*(a1 + a2) ----
    k_combine<<<cdiv((long)FMN, 256), 256, 0, stream>>>(xa1, fm_chan, (float*)d_out,
                                                        (long)FMN, 196, 768);
}
